// RetNetEncoderLayer_3161095930392
// MI455X (gfx1250) — matrix operand order repaired and checked by reading, not compiled, not run on hardware
//
#include <hip/hip_runtime.h>
#include <hip/hip_bf16.h>

// ---------------- constants (RetNet layer, B=2,S=2048,D=1024,H=8) ----------
#define BB   2
#define SS   2048
#define DD   1024
#define HH   8
#define DK   128
#define VD   2048
#define DV   256
#define FFN  4096
#define MM   (BB*SS)          // 4096 rows

typedef __bf16 bf16;
typedef bf16  v16bf __attribute__((ext_vector_type(16)));
typedef bf16  v8bf  __attribute__((ext_vector_type(8)));
typedef float v8f   __attribute__((ext_vector_type(8)));
typedef int   v4i   __attribute__((ext_vector_type(4)));
typedef int   v8i   __attribute__((ext_vector_type(8)));

__device__ __forceinline__ bf16 f2bf(float f) {
  unsigned u = __builtin_bit_cast(unsigned, f);
  u += 0x7FFFu + ((u >> 16) & 1u);                 // round-to-nearest-even
  unsigned short h = (unsigned short)(u >> 16);
  return __builtin_bit_cast(bf16, h);
}

// Build a 16-wide bf16 WMMA fragment from two 16-byte LDS/global chunks.
__device__ __forceinline__ v16bf load_frag16(const bf16* p0, const bf16* p1) {
  v8bf a = *(const v8bf*)p0;
  v8bf b = *(const v8bf*)p1;
  v16bf r;
#pragma unroll
  for (int i = 0; i < 8; ++i) { r[i] = a[i]; r[i + 8] = b[i]; }
  return r;
}

// ---------------- Tensor Data Mover: 2D bf16 tile -> padded LDS ------------
// Loads `rows` x `cols_halves` bf16 tile (global row stride = stride_halves)
// into LDS at lds_off, inserting pad_amount(+1) DWORDs of pad every
// 2<<pad_interval_code DWORDs (reproduces the padded LDS row stride).
// D# per CDNA5 ISA ch.8: group0 = {count/lds_addr/global_addr/type},
// group1 = {mask,data_size,pad,dims,strides}.  Issued per-wave; completion
// tracked with TENSORcnt.
__device__ __forceinline__ void tdm_load_2d(const bf16* gptr, unsigned lds_off,
                                            int rows, int cols_halves,
                                            int stride_halves, int tensor_rows,
                                            int pad_interval_code, int pad_amount_code) {
  unsigned long long ga = (unsigned long long)(const void*)gptr;
  v4i g0;
  g0[0] = 1;                                            // count=1, user mode
  g0[1] = (int)lds_off;                                 // lds_addr (bytes)
  g0[2] = (int)(unsigned)(ga & 0xFFFFFFFFu);            // global_addr[31:0]
  g0[3] = (int)(((unsigned)(ga >> 32) & 0x01FFFFFFu) | 0x80000000u); // addr[56:32] | type=2
  v8i g1;
  unsigned dw0 = (1u << 16)                             // data_size = 2 bytes
               | (1u << 20)                             // pad_enable
               | ((unsigned)pad_interval_code << 22)
               | ((unsigned)pad_amount_code << 25);
  g1[0] = (int)dw0;                                     // workgroup_mask=0 (not in cluster)
  g1[1] = (int)((unsigned)(stride_halves & 0xFFFF) << 16);            // tensor_dim0[15:0]
  g1[2] = (int)(((unsigned)(stride_halves >> 16) & 0xFFFFu)           // tensor_dim0[31:16]
              | (((unsigned)tensor_rows & 0xFFFFu) << 16));           // tensor_dim1[15:0]
  g1[3] = (int)((((unsigned)tensor_rows >> 16) & 0xFFFFu)             // tensor_dim1[31:16]
              | (((unsigned)cols_halves & 0xFFFFu) << 16));           // tile_dim0
  g1[4] = rows & 0xFFFF;                                // tile_dim1 (tile_dim2 = 0)
  g1[5] = stride_halves;                                // tensor_dim0_stride[31:0]
  g1[6] = 0;                                            // stride high bits / dim1_stride
  g1[7] = 0;
  asm volatile("tensor_load_to_lds %0, %1" :: "s"(g0), "s"(g1) : "memory");
}

// ---------------- weight conversion / packing ------------------------------
__global__ __launch_bounds__(256) void cvt_bf16_kernel(const float* __restrict__ in,
                                                       bf16* __restrict__ out, int n) {
  int i = blockIdx.x * 256 + threadIdx.x;
  if (i < n) out[i] = f2bf(in[i]);
}

// (H, D, Kd) -> (D, H*Kd) bf16
__global__ __launch_bounds__(256) void pack_hdk_kernel(const float* __restrict__ in,
                                                       bf16* __restrict__ out,
                                                       int H_, int D_, int Kd) {
  int i = blockIdx.x * 256 + threadIdx.x;
  int total = H_ * D_ * Kd;
  if (i >= total) return;
  int k = i % Kd; int t = i / Kd; int d = t % D_; int h = t / D_;
  out[(size_t)d * (H_ * Kd) + h * Kd + k] = f2bf(in[i]);
}

// ---------------- layernorm -> bf16 ----------------------------------------
__global__ __launch_bounds__(256) void layernorm_bf16_kernel(const float* __restrict__ X,
                                                             const float* __restrict__ w,
                                                             const float* __restrict__ b,
                                                             bf16* __restrict__ out, int D) {
  __shared__ float red[256];
  const int row = blockIdx.x;
  const float* x = X + (size_t)row * D;
  float s = 0.f, ss = 0.f;
  for (int i = threadIdx.x; i < D; i += 256) { float v = x[i]; s += v; ss += v * v; }
  red[threadIdx.x] = s; __syncthreads();
  for (int o = 128; o > 0; o >>= 1) { if (threadIdx.x < o) red[threadIdx.x] += red[threadIdx.x + o]; __syncthreads(); }
  float mean = red[0] / (float)D; __syncthreads();
  red[threadIdx.x] = ss; __syncthreads();
  for (int o = 128; o > 0; o >>= 1) { if (threadIdx.x < o) red[threadIdx.x] += red[threadIdx.x + o]; __syncthreads(); }
  float var = red[0] / (float)D - mean * mean;
  float inv = rsqrtf(var + 1e-5f);
  for (int i = threadIdx.x; i < D; i += 256)
    out[(size_t)row * D + i] = f2bf((x[i] - mean) * inv * w[i] + b[i]);
}

// ---------------- generic bf16 WMMA GEMM -----------------------------------
// C(MxN) = A(MxK) @ B(KxN); 64x64 block, 4 waves, each wave 32x32 (2x2 WMMA).
// A tile staged to LDS by the Tensor Data Mover (wave 0), B tile transposed
// manually by all waves, overlapped with the TDM transfer.
// EPI: 0 f32 store | 1 bf16 store | 2 bias+GeLU->bf16 | 3 bias+resid->f32 | 4 resid->f32
#define LDT 40     // LDS row stride in halves (80B, 16B aligned): 64B row + 16B TDM pad

template <int EPI>
__global__ __launch_bounds__(128)
void gemm_bf16_kernel(const bf16* __restrict__ A, const bf16* __restrict__ Bm,
                      const float* __restrict__ bias, const float* __restrict__ resid,
                      void* __restrict__ Cout, int M, int N, int K) {
  __shared__ bf16 sA[64 * LDT];
  __shared__ bf16 sBT[64 * LDT];   // N-major (transposed) B tile
  const int tid = threadIdx.x;
  const int lane = tid & 31, wave = tid >> 5;
  const int lr = lane & 15, lh = lane >> 4;
  const int m0 = blockIdx.y * 64, n0 = blockIdx.x * 64;
  const int wm = (wave >> 1) * 32, wn = (wave & 1) * 32;
  const unsigned sA_off = (unsigned)(size_t)(&sA[0]);   // LDS byte offset (addr[31:0])

  v8f acc[2][2] = {};
  for (int k0 = 0; k0 < K; k0 += 32) {
    __syncthreads();
    if (wave == 0)                                      // TDM: A tile 64x32 -> padded LDS
      tdm_load_2d(A + (size_t)m0 * K + k0, sA_off, 64, 32, K, M - m0,
                  /*interval 16 DW = 64B*/ 3, /*pad 4 DW = 16B*/ 3);
#pragma unroll
    for (int i = 0; i < 4; ++i) {                       // B tile 32x64, transpose
      int cid = tid + i * 128;
      int kk = cid >> 4, nc = (cid & 15) * 4;
      uint2 v = *(const uint2*)(Bm + (size_t)(k0 + kk) * N + n0 + nc);
      const bf16* pv = (const bf16*)&v;
#pragma unroll
      for (int j = 0; j < 4; ++j) sBT[(nc + j) * LDT + kk] = pv[j];
    }
    if (k0 + 32 < K)                                    // hint next B tile
      __builtin_prefetch(Bm + (size_t)(k0 + 32 + (tid & 31)) * N + n0, 0, 0);
    if (wave == 0) __builtin_amdgcn_s_wait_tensorcnt(0);
    __syncthreads();
    v16bf af[2], bfr[2];
#pragma unroll
    for (int i = 0; i < 2; ++i) {
      const bf16* p = &sA[(wm + i * 16 + lr) * LDT + lh * 8];
      af[i] = load_frag16(p, p + 16);
    }
#pragma unroll
    for (int j = 0; j < 2; ++j) {
      const bf16* p = &sBT[(wn + j * 16 + lr) * LDT + lh * 8];
      bfr[j] = load_frag16(p, p + 16);
    }
#pragma unroll
    for (int i = 0; i < 2; ++i)
#pragma unroll
      for (int j = 0; j < 2; ++j)
        acc[i][j] = __builtin_amdgcn_wmma_f32_16x16x32_bf16(
            false, af[i], false, bfr[j], (short)0, acc[i][j], false, false);
  }
#pragma unroll
  for (int i = 0; i < 2; ++i)
#pragma unroll
    for (int j = 0; j < 2; ++j)
#pragma unroll
      for (int r = 0; r < 8; ++r) {
        int m = m0 + wm + i * 16 + lh * 8 + r;
        int n = n0 + wn + j * 16 + lr;
        size_t idx = (size_t)m * N + n;
        float v = acc[i][j][r];
        if constexpr (EPI == 0) ((float*)Cout)[idx] = v;
        else if constexpr (EPI == 1) ((bf16*)Cout)[idx] = f2bf(v);
        else if constexpr (EPI == 2) {
          v += bias[n];
          v = 0.5f * v * (1.f + erff(v * 0.70710678f));
          ((bf16*)Cout)[idx] = f2bf(v);
        } else if constexpr (EPI == 3) ((float*)Cout)[idx] = v + bias[n] + resid[idx];
        else ((float*)Cout)[idx] = v + resid[idx];
      }
}

// ---------------- xPos rotary: f32 (BS, H*DK) -> bf16 (B,H,S,DK) -----------
__global__ __launch_bounds__(64) void xpos_kernel(const float* __restrict__ Qr,
                                                  bf16* __restrict__ Qb, int downscale) {
  const int m = blockIdx.x;             // b*S + s
  const int h = blockIdx.y;
  const int i = threadIdx.x;            // pair index 0..63
  const int s = m & (SS - 1);
  const int b = m >> 11;
  const float* src = Qr + (size_t)m * (HH * DK) + h * DK;
  float x0 = src[2 * i], x1 = src[2 * i + 1];
  float pos  = (float)s;
  float invf = powf(10000.f, -(float)i / 64.f);
  float sv   = (2.f * (float)i + 0.4f * (float)DK) / (1.4f * (float)DK);
  float e    = pos * (1.f / 512.f);
  float scale = powf(sv, downscale ? -e : e);
  float ang = pos * invf;
  float cn = cosf(ang) * scale, sn = sinf(ang) * scale;
  bf16* dst = Qb + ((size_t)(b * HH + h) * SS + s) * DK;
  dst[2 * i]     = f2bf(x0 * cn - x1 * sn);
  dst[2 * i + 1] = f2bf(x1 * cn + x0 * sn);
}

// ---------------- fused retention (flash-style, WMMA both stages) ----------
// grid: (S/64, B*H); 4 waves; wave w owns q rows [q0+16w, q0+16w+16).
// K tile staged by TDM (wave 0), V transpose overlapped on all waves.
__global__ __launch_bounds__(128)
void retention_kernel(const bf16* __restrict__ Qb, const bf16* __restrict__ Kb,
                      const bf16* __restrict__ Vb, float* __restrict__ Y) {
  __shared__ bf16 sK[32 * 136];        // key tile, row-major [t][dk]; 256B row + 16B pad
  __shared__ bf16 sVT[DV * LDT];       // value tile, transposed [dv][t]
  __shared__ bf16 sS[4 * 16 * LDT];    // per-wave score tiles [q][t]
  const int tid = threadIdx.x, lane = tid & 31, wave = tid >> 5;
  const int lr = lane & 15, lh = lane >> 4;
  const int bh = blockIdx.y, b = bh >> 3, h = bh & 7;
  const int q0 = blockIdx.x * 64;
  const int qrow = q0 + wave * 16;
  const unsigned sK_off = (unsigned)(size_t)(&sK[0]);

  const float lg1 = -3.4657359028f, lg2 = -6.2383246250f;   // log(1/32), log(1/512)
  const float gam = 1.f - expf(lg1 + (float)h * (lg2 - lg1) / 7.f);
  const float l2g = log2f(gam);

  const bf16* qp = Qb + ((size_t)bh * SS + (qrow + lr)) * DK;
  v16bf qf[4];
#pragma unroll
  for (int kk = 0; kk < 4; ++kk) {
    const bf16* p = qp + kk * 32 + lh * 8;
    qf[kk] = load_frag16(p, p + 16);
  }

  v8f yacc[16] = {};
  bf16* sSw = &sS[wave * 16 * LDT];

  for (int t0 = 0; t0 < q0 + 64; t0 += 32) {
    __syncthreads();
    if (wave == 0)                                      // TDM: K tile 32x128 -> padded LDS
      tdm_load_2d(Kb + ((size_t)bh * SS + t0) * DK, sK_off, 32, DK, DK, SS - t0,
                  /*interval 64 DW = 256B*/ 5, /*pad 4 DW = 16B*/ 3);
#pragma unroll
    for (int i = 0; i < 16; ++i) {                      // V tile 32x256, transpose
      int cid = tid + i * 128;
      int t = cid >> 6, dc = (cid & 63) * 4;
      uint2 v = *(const uint2*)(Vb + (size_t)(b * SS + t0 + t) * VD + h * DV + dc);
      const bf16* pv = (const bf16*)&v;
#pragma unroll
      for (int j = 0; j < 4; ++j) sVT[(dc + j) * LDT + t] = pv[j];
    }
    if (wave == 0) __builtin_amdgcn_s_wait_tensorcnt(0);
    __syncthreads();

    v8f sacc[2] = {};
#pragma unroll
    for (int kk = 0; kk < 4; ++kk)                      // stage 1: Q K^T
#pragma unroll
      for (int j = 0; j < 2; ++j) {
        const bf16* p = &sK[(j * 16 + lr) * 136 + kk * 32 + lh * 8];
        v16bf bfr = load_frag16(p, p + 16);
        sacc[j] = __builtin_amdgcn_wmma_f32_16x16x32_bf16(
            false, qf[kk], false, bfr, (short)0, sacc[j], false, false);
      }
    // decay mask + requantize scores into per-wave LDS
#pragma unroll
    for (int j = 0; j < 2; ++j)
#pragma unroll
      for (int r = 0; r < 8; ++r) {
        int srow = qrow + lh * 8 + r;
        int tt = t0 + j * 16 + lr;
        int d = srow - tt;
        float v = sacc[j][r];
        v = (d >= 0) ? v * exp2f((float)d * l2g) : 0.f;
        sSw[(lh * 8 + r) * LDT + j * 16 + lr] = f2bf(v);
      }
    __syncthreads();
    const bf16* sp = &sSw[lr * LDT + lh * 8];
    v16bf sfrag = load_frag16(sp, sp + 16);
#pragma unroll
    for (int n = 0; n < 16; ++n) {                      // stage 2: ret @ V
      const bf16* p = &sVT[(n * 16 + lr) * LDT + lh * 8];
      v16bf vfr = load_frag16(p, p + 16);
      yacc[n] = __builtin_amdgcn_wmma_f32_16x16x32_bf16(
          false, sfrag, false, vfr, (short)0, yacc[n], false, false);
    }
  }
#pragma unroll
  for (int n = 0; n < 16; ++n)
#pragma unroll
    for (int r = 0; r < 8; ++r) {
      int srow = qrow + lh * 8 + r;
      int dv = n * 16 + lr;
      Y[(size_t)(b * SS + srow) * VD + h * DV + dv] = yacc[n][r];
    }
}

// ---------------- groupnorm + SiLU gate ------------------------------------
__global__ __launch_bounds__(256) void gn_gate_kernel(const float* __restrict__ Y,
                                                      const float* __restrict__ G,
                                                      const float* __restrict__ gw,
                                                      const float* __restrict__ gb,
                                                      bf16* __restrict__ out) {
  __shared__ float red[256];
  const int g = blockIdx.x;             // bs*H + h
  const int bs = g >> 3, h = g & 7;
  const int dv = threadIdx.x;
  const size_t idx = (size_t)bs * VD + h * DV + dv;
  const float y = Y[idx];
  red[dv] = y; __syncthreads();
  for (int o = 128; o > 0; o >>= 1) { if (dv < o) red[dv] += red[dv + o]; __syncthreads(); }
  float mean = red[0] * (1.f / 256.f); __syncthreads();
  red[dv] = y * y; __syncthreads();
  for (int o = 128; o > 0; o >>= 1) { if (dv < o) red[dv] += red[dv + o]; __syncthreads(); }
  float var = red[0] * (1.f / 256.f) - mean * mean;
  float yn = (y - mean) * rsqrtf(var + 1e-5f) * gw[h * DV + dv] + gb[h * DV + dv];
  float gv = G[idx];
  float sg = 1.f / (1.f + expf(-gv));
  out[idx] = f2bf(gv * sg * yn);
}

// ---------------- host-side orchestration ----------------------------------
static inline size_t align_up(size_t x) { return (x + 255) & ~(size_t)255; }

extern "C" void kernel_launch(void* const* d_in, const int* in_sizes, int n_in,
                              void* d_out, int out_size, void* d_ws, size_t ws_size,
                              hipStream_t stream) {
  const float* X    = (const float*)d_in[0];
  const float* Wq   = (const float*)d_in[1];
  const float* Wk   = (const float*)d_in[2];
  const float* Wv   = (const float*)d_in[3];
  const float* W_G  = (const float*)d_in[4];
  const float* W_O  = (const float*)d_in[5];
  const float* gn_w = (const float*)d_in[6];
  const float* gn_b = (const float*)d_in[7];
  const float* ln1w = (const float*)d_in[8];
  const float* ln1b = (const float*)d_in[9];
  const float* ln2w = (const float*)d_in[10];
  const float* ln2b = (const float*)d_in[11];
  const float* W1   = (const float*)d_in[12];
  const float* b1   = (const float*)d_in[13];
  const float* W2   = (const float*)d_in[14];
  const float* b2   = (const float*)d_in[15];

  char* w = (char*)d_ws;
  auto alloc = [&](size_t bytes) { char* p = w; w += align_up(bytes); return p; };

  bf16*  XnB   = (bf16*) alloc((size_t)MM * DD * 2);
  float* Qraw  = (float*)alloc((size_t)MM * DD * 4);
  float* Kraw  = (float*)alloc((size_t)MM * DD * 4);
  bf16*  Qb    = (bf16*) alloc((size_t)MM * DD * 2);
  bf16*  Kb    = (bf16*) alloc((size_t)MM * DD * 2);
  bf16*  Vb    = (bf16*) alloc((size_t)MM * VD * 2);
  float* Gbuf  = (float*)alloc((size_t)MM * VD * 4);
  float* Ybuf  = (float*)alloc((size_t)MM * VD * 4);
  bf16*  gated = (bf16*) alloc((size_t)MM * VD * 2);
  float* X2    = (float*)alloc((size_t)MM * DD * 4);
  bf16*  X2nB  = (bf16*) alloc((size_t)MM * DD * 2);
  bf16*  H1B   = (bf16*) alloc((size_t)MM * FFN * 2);
  bf16*  WqP   = (bf16*) alloc((size_t)DD * DD * 2);
  bf16*  WkP   = (bf16*) alloc((size_t)DD * DD * 2);
  bf16*  WvP   = (bf16*) alloc((size_t)DD * VD * 2);
  bf16*  WgB   = (bf16*) alloc((size_t)DD * VD * 2);
  bf16*  WoB   = (bf16*) alloc((size_t)VD * DD * 2);
  bf16*  W1B   = (bf16*) alloc((size_t)DD * FFN * 2);
  bf16*  W2B   = (bf16*) alloc((size_t)FFN * DD * 2);

  // 1) weight conversion / packing
  pack_hdk_kernel<<<(HH*DD*DK + 255)/256, 256, 0, stream>>>(Wq, WqP, HH, DD, DK);
  pack_hdk_kernel<<<(HH*DD*DK + 255)/256, 256, 0, stream>>>(Wk, WkP, HH, DD, DK);
  pack_hdk_kernel<<<(HH*DD*DV + 255)/256, 256, 0, stream>>>(Wv, WvP, HH, DD, DV);
  cvt_bf16_kernel<<<(DD*VD + 255)/256, 256, 0, stream>>>(W_G, WgB, DD*VD);
  cvt_bf16_kernel<<<(VD*DD + 255)/256, 256, 0, stream>>>(W_O, WoB, VD*DD);
  cvt_bf16_kernel<<<(DD*FFN + 255)/256, 256, 0, stream>>>(W1, W1B, DD*FFN);
  cvt_bf16_kernel<<<(FFN*DD + 255)/256, 256, 0, stream>>>(W2, W2B, FFN*DD);

  // 2) LN1 -> bf16 activations
  layernorm_bf16_kernel<<<MM, 256, 0, stream>>>(X, ln1w, ln1b, XnB, DD);

  // 3) projections (WMMA GEMMs, TDM-staged A tiles)
  dim3 blk(128);
  gemm_bf16_kernel<0><<<dim3(DD/64,  MM/64), blk, 0, stream>>>(XnB, WqP, nullptr, nullptr, Qraw, MM, DD,  DD);
  gemm_bf16_kernel<0><<<dim3(DD/64,  MM/64), blk, 0, stream>>>(XnB, WkP, nullptr, nullptr, Kraw, MM, DD,  DD);
  gemm_bf16_kernel<1><<<dim3(VD/64,  MM/64), blk, 0, stream>>>(XnB, WvP, nullptr, nullptr, Vb,   MM, VD,  DD);
  gemm_bf16_kernel<0><<<dim3(VD/64,  MM/64), blk, 0, stream>>>(XnB, WgB, nullptr, nullptr, Gbuf, MM, VD,  DD);

  // 4) xPos rotary (Q up-scale, K down-scale) + head-major repack
  xpos_kernel<<<dim3(MM, HH), 64, 0, stream>>>(Qraw, Qb, 0);
  xpos_kernel<<<dim3(MM, HH), 64, 0, stream>>>(Kraw, Kb, 1);

  // 5) fused decay-masked retention
  retention_kernel<<<dim3(SS/64, BB*HH), blk, 0, stream>>>(Qb, Kb, Vb, Ybuf);

  // 6) groupnorm + SiLU gating
  gn_gate_kernel<<<MM*HH, 256, 0, stream>>>(Ybuf, Gbuf, gn_w, gn_b, gated);

  // 7) output projection + residual
  gemm_bf16_kernel<4><<<dim3(DD/64, MM/64), blk, 0, stream>>>(gated, WoB, nullptr, X, X2, MM, DD, VD);

  // 8) LN2, FFN with exact GeLU, second residual -> d_out (fp32)
  layernorm_bf16_kernel<<<MM, 256, 0, stream>>>(X2, ln2w, ln2b, X2nB, DD);
  gemm_bf16_kernel<2><<<dim3(FFN/64, MM/64), blk, 0, stream>>>(X2nB, W1B, b1, nullptr, H1B, MM, FFN, DD);
  gemm_bf16_kernel<3><<<dim3(DD/64,  MM/64), blk, 0, stream>>>(H1B, W2B, b2, X2, (float*)d_out, MM, DD, FFN);
}
